// HarveySVPF_58076547777185
// MI455X (gfx1250) — matrix operand order repaired and checked by reading, not compiled, not run on hardware
//
#include <hip/hip_runtime.h>
#include <hip/hip_bf16.h>
#include <math.h>

// ---- problem constants (match reference) ----
#define N_PART   256
#define T_STEPS  1024
#define BATCH    256
#define PF_ALPHA 0.3f

typedef __attribute__((ext_vector_type(16))) _Float16 v16h;
typedef __attribute__((ext_vector_type(8)))  float    v8f;

// ---- deterministic counter-based RNG (PCG output hash) ----
__device__ __forceinline__ unsigned pcg_hash(unsigned x) {
    x = x * 747796405u + 2891336453u;
    unsigned w = ((x >> ((x >> 28u) + 4u)) ^ x) * 277803737u;
    return (w >> 22u) ^ w;
}
__device__ __forceinline__ float u01(unsigned s) {          // [0,1)
    return (float)(pcg_hash(s) >> 8) * (1.0f / 16777216.0f);
}
__device__ __forceinline__ float gauss(unsigned s) {        // Box-Muller
    unsigned ua = pcg_hash(s ^ 0x68bc21ebu);
    unsigned ub = pcg_hash(s ^ 0x02e5be93u);
    float u1 = (float)((ua >> 8) + 1u) * (1.0f / 16777216.0f);  // (0,1]
    float u2 = (float)(ub >> 8) * (1.0f / 16777216.0f);
    return sqrtf(-2.0f * __logf(u1)) * __cosf(6.28318530718f * u2);
}

// ---- wave32 shuffle collectives (no barriers) ----
__device__ __forceinline__ float waveSum(float v) {
#pragma unroll
    for (int off = 16; off > 0; off >>= 1) v += __shfl_xor(v, off, 32);
    return v;
}
__device__ __forceinline__ float waveMax(float v) {
#pragma unroll
    for (int off = 16; off > 0; off >>= 1) v = fmaxf(v, __shfl_xor(v, off, 32));
    return v;
}
__device__ __forceinline__ float waveInclScan(float v, int lane) {
#pragma unroll
    for (int off = 1; off < 32; off <<= 1) {
        float n = __shfl_up(v, off, 32);
        if (lane >= off) v += n;
    }
    return v;
}

// One block per batch element b; thread n = particle n. Whole T-scan fused.
// 8 waves/block; cross-wave combines use 8-float LDS arrays, 1 barrier each.
__global__ __launch_bounds__(N_PART)
void pf_scan_kernel(const float* __restrict__ obs,   // [B,T,1]
                    const float* __restrict__ Wz, const float* __restrict__ bz,
                    const float* __restrict__ Wr, const float* __restrict__ br,
                    const float* __restrict__ Wn, const float* __restrict__ bn,
                    const float* __restrict__ Wo, const float* __restrict__ bo,
                    float* __restrict__ out_y,       // [T,B]
                    float* __restrict__ out_h)       // [T,N*B]
{
    const int b    = blockIdx.x;
    const int tid  = threadIdx.x;
    const int lane = tid & 31;          // wave32
    const int wave = tid >> 5;          // 8 waves / block

    __shared__ float sH[N_PART];        // h1, then h_new
    __shared__ float sP[N_PART];        // p1, then p_new
    __shared__ float sCdf[N_PART];      // resampling CDF
    __shared__ float sWMax[8], sWSum[8], sWScan[8], sWNorm[8], sWY[8];

    const float wz_h = Wz[0], wz_x = Wz[1], bz0 = bz[0];
    const float wr_h = Wr[0], wr_x = Wr[1], br0 = br[0];
    const float wn00 = Wn[0], wn01 = Wn[1], wn10 = Wn[2], wn11 = Wn[3];
    const float bn0  = bn[0], bn1  = bn[1];
    const float wo_h = Wo[0], wo_x = Wo[1], bo0 = bo[0];

    // init_hidden: h ~ U[-2,2), p = 1/N
    float h = u01(((unsigned)tid << 16) ^ (unsigned)b ^ 0x42424242u) * 4.0f - 2.0f;
    float p = 1.0f / (float)N_PART;

    const float* obs_b = obs + (size_t)b * T_STEPS;   // IN_DIM == 1

    for (int t = 0; t < T_STEPS; ++t) {
        if (t + 16 < T_STEPS)
            __builtin_prefetch(obs_b + t + 16, 0, 1); // global_prefetch_b8
        const float x = obs_b[t];

        // ---- per-particle GRU-SSM transition ----
        float z  = 1.0f / (1.0f + __expf(-(wz_h * h + wz_x * x + bz0)));
        float r  = 1.0f / (1.0f + __expf(-(wr_h * h + wr_x * x + br0)));
        float rh = r * h;
        float mu  = wn00 * rh + wn01 * x + bn0;
        float var = wn10 * rh + wn11 * x + bn1;
        float stdv = (var > 20.0f) ? var : log1pf(__expf(var));   // softplus
        unsigned st = (unsigned)t * 0x9E3779B1u
                    ^ (unsigned)tid * 0x85EBCA77u
                    ^ (unsigned)b * 0xC2B2AE3Du;
        float nval = tanhf(mu + stdv * gauss(st));
        float h1   = (1.0f - z) * nval + z * h;
        float logw = wo_h * h1 + wo_x * x + bo0;

        // ---- log-softmax over N=256 particles: wave reduce + 1-barrier join ----
        float lp = __logf(p) + logw;
        float wm = waveMax(lp);
        if (lane == 0) sWMax[wave] = wm;
        __syncthreads();                                     // barrier 1
        float M = sWMax[0];
#pragma unroll
        for (int w = 1; w < 8; ++w) M = fmaxf(M, sWMax[w]);

        float e  = __expf(lp - M);
        float ws = waveSum(e);
        if (lane == 0) sWSum[wave] = ws;
        // publish h1/p1 under the next barrier
        sH[tid] = h1;
        __syncthreads();                                     // barrier 2
        float S = 0.0f;
#pragma unroll
        for (int w = 0; w < 8; ++w) S += sWSum[w];
        float p1 = e / S;
        sP[tid] = p1;

        // ---- soft-resampling CDF: wave scan + cross-wave offset ----
        float q  = PF_ALPHA * p1 + (1.0f - PF_ALPHA) / (float)N_PART;
        float sc = waveInclScan(q, lane);
        if (lane == 31) sWScan[wave] = sc;                   // wave totals
        __syncthreads();                                     // barrier 3 (also fences sP)
        float off = 0.0f;
#pragma unroll
        for (int w = 0; w < 8; ++w) if (w < wave) off += sWScan[w];
        sCdf[tid] = sc + off;
        __syncthreads();                                     // barrier 4
        float total = sCdf[N_PART - 1];

        // ---- categorical sample + gather ----
        float u = u01(st ^ 0x7feb352du) * total;
        int lo = 0, hi = N_PART - 1;
        while (lo < hi) {               // 8-step binary search in LDS CDF
            int mid = (lo + hi) >> 1;
            if (sCdf[mid] > u) hi = mid; else lo = mid + 1;
        }
        float h_new = sH[lo];
        float p_g   = sP[lo];
        float pn_un = p_g / (PF_ALPHA * p_g + (1.0f - PF_ALPHA) / (float)N_PART);

        float wn = waveSum(pn_un);
        if (lane == 0) sWNorm[wave] = wn;
        __syncthreads();                                     // barrier 5 (fences gathers)
        float norm = 0.0f;
#pragma unroll
        for (int w = 0; w < 8; ++w) norm += sWNorm[w];
        float p_new = pn_un / norm;

        // publish resampled state for the WMMA weighted-mean
        sH[tid] = h_new;
        sP[tid] = p_new;
        __syncthreads();                                     // barrier 6

        // particle trajectory output: pf_out[t, n, b] (write-once -> NT)
        __builtin_nontemporal_store(
            h_new, out_h + (size_t)t * (N_PART * BATCH) + (size_t)tid * BATCH + b);

        // ---- y[t,b] = sum_n h_new*p_new via v_wmma_f32_16x16x32_f16 ----
        // Each wave: 32-element partial dot. A row M=0 carries the h chunk
        // (ISA 16-bit A striping), B carries the p chunk in every column;
        // D[0][0] (VGPR0 @ lane 0) is the partial.
        {
            const int base = wave * 32;
            v16h av, bv;
#pragma unroll
            for (int i = 0; i < 16; ++i) av[i] = (_Float16)0.0f;
            if (lane == 0) {            // M=0: K = 0..7, 16..23
#pragma unroll
                for (int i = 0; i < 8; ++i) {
                    av[i]     = (_Float16)sH[base + i];
                    av[8 + i] = (_Float16)sH[base + 16 + i];
                }
            } else if (lane == 16) {    // M=0: K = 8..15, 24..31
#pragma unroll
                for (int i = 0; i < 8; ++i) {
                    av[i]     = (_Float16)sH[base + 8 + i];
                    av[8 + i] = (_Float16)sH[base + 24 + i];
                }
            }
            {
                const int koff = (lane < 16) ? 0 : 8;   // B: K = koff+0..7, 16+koff+0..7
#pragma unroll
                for (int i = 0; i < 16; ++i)
                    bv[i] = (_Float16)sP[base + koff + i + (i & 8)];
            }
            v8f c = {};
            c = __builtin_amdgcn_wmma_f32_16x16x32_f16(
                    false, av, false, bv, (short)0, c, false, false);
            if (lane == 0) sWY[wave] = c[0];    // D[0][0]
        }
        __syncthreads();                                     // barrier 7
        if (tid == 0) {
            float y = 0.0f;
#pragma unroll
            for (int w = 0; w < 8; ++w) y += sWY[w];
            __builtin_nontemporal_store(y, out_y + (size_t)t * BATCH + b);
        }

        h = h_new;
        p = p_new;
    }
}

extern "C" void kernel_launch(void* const* d_in, const int* in_sizes, int n_in,
                              void* d_out, int out_size, void* d_ws, size_t ws_size,
                              hipStream_t stream) {
    (void)in_sizes; (void)n_in; (void)out_size; (void)d_ws; (void)ws_size;
    const float* obs = (const float*)d_in[0];
    const float* Wz  = (const float*)d_in[1];
    const float* bz  = (const float*)d_in[2];
    const float* Wr  = (const float*)d_in[3];
    const float* br  = (const float*)d_in[4];
    const float* Wn  = (const float*)d_in[5];
    const float* bn  = (const float*)d_in[6];
    const float* Wo  = (const float*)d_in[7];
    const float* bo  = (const float*)d_in[8];

    float* out   = (float*)d_out;
    float* out_y = out;                                   // T*B floats
    float* out_h = out + (size_t)T_STEPS * BATCH;         // T*N*B floats

    hipLaunchKernelGGL(pf_scan_kernel, dim3(BATCH), dim3(N_PART), 0, stream,
                       obs, Wz, bz, Wr, br, Wn, bn, Wo, bo, out_y, out_h);
}
